// GraphAutoEncoder_46420006535376
// MI455X (gfx1250) — matrix operand order repaired and checked by reading, not compiled, not run on hardware
//
#include <hip/hip_runtime.h>
#include <hip/hip_bf16.h>

// ---------------- problem constants ----------------
#define N_NODES 10000
#define SEQ     32
#define PDIM    512
#define ENCD    256
#define FFD     1024
#define KNB     32
#define HEADS   4
#define DH      64

typedef __attribute__((ext_vector_type(16))) __bf16       v16bf;
typedef __attribute__((ext_vector_type(8)))  float        v8f;
typedef __attribute__((ext_vector_type(8)))  unsigned int v8u;

__device__ __forceinline__ unsigned short f2bf(float f) {
  unsigned int u = __builtin_bit_cast(unsigned int, f);
  u += 0x7fffu + ((u >> 16) & 1u);          // round-to-nearest-even
  return (unsigned short)(u >> 16);
}

// Build one WMMA 16-element bf16 fragment from two contiguous 16B chunks.
__device__ __forceinline__ v16bf load_frag(const unsigned short* p0,
                                           const unsigned short* p1) {
  uint4 a = *(const uint4*)p0;
  uint4 b = *(const uint4*)p1;
  v8u r;
  r[0]=a.x; r[1]=a.y; r[2]=a.z; r[3]=a.w;
  r[4]=b.x; r[5]=b.y; r[6]=b.z; r[7]=b.w;
  return __builtin_bit_cast(v16bf, r);
}

__device__ __forceinline__ v8f wmma_bf16(v16bf a, v16bf b, v8f c) {
  return __builtin_amdgcn_wmma_f32_16x16x32_bf16(false, a, false, b,
                                                 (short)0, c, false, false);
}

// Generic tiled GEMM:  C[M,N] = A[M,K] * BT[N,K]^T  (+bias, optional relu)
// A: bf16 bits (LDS), BT: bf16 bits [N][K] (global, L2-resident), 8 waves.
template<int KDIM>
__device__ __forceinline__ void gemm_bf16(const unsigned short* A, int lda,
                                          const unsigned short* BT, int ldb,
                                          const float* bias,
                                          float* Cf, int ldcf,
                                          unsigned short* Cb, int ldcb,
                                          int Mtiles, int Ntiles,
                                          bool relu, int tid) {
  const int lane = tid & 31, wave = tid >> 5;
  const int khalfA = (lane & 16) ? 8 : 0;     // A-frag K sub-offset per lane half
  const int kloB   = (lane & 16) ? 8 : 0;     // B-frag K sub-offset per lane half
  const int total  = Mtiles * Ntiles;
  for (int t = wave; t < total; t += 8) {
    const int mt = t % Mtiles, nt = t / Mtiles;
    const int m  = mt * 16 + (lane & 15);
    const int nB = nt * 16 + (lane & 15);
    v8f c = {};
    for (int k0 = 0; k0 < KDIM; k0 += 32) {
      v16bf a = load_frag(A + m * lda + k0 + khalfA,
                          A + m * lda + k0 + 16 + khalfA);
      v16bf b = load_frag(BT + nB * ldb + k0 + kloB,
                          BT + nB * ldb + k0 + 16 + kloB);
      c = wmma_bf16(a, b, c);
    }
    const int n    = nt * 16 + (lane & 15);
    const float bb = bias ? bias[n] : 0.f;
    const int mr0  = mt * 16 + ((lane & 16) ? 8 : 0);
    for (int r = 0; r < 8; ++r) {
      float v = c[r] + bb;
      if (relu) v = fmaxf(v, 0.f);
      if (Cf) Cf[(mr0 + r) * ldcf + n] = v;
      if (Cb) Cb[(mr0 + r) * ldcb + n] = f2bf(v);
    }
  }
}

// residual + LayerNorm over 32 rows x 256 cols; 8 threads per row.
__device__ __forceinline__ void ln_residual(float* xres, const float* add,
                                            const float* g, const float* b,
                                            unsigned short* xb, int tid) {
  const int row = tid >> 3, j = tid & 7;       // 8-lane groups live in one wave
  float vals[32];
  float s = 0.f;
  for (int c = 0; c < 32; ++c) {
    float v = xres[row * ENCD + j * 32 + c] + add[row * ENCD + j * 32 + c];
    vals[c] = v; s += v;
  }
  for (int m = 1; m < 8; m <<= 1) s += __shfl_xor(s, m, 32);
  const float mu = s * (1.f / ENCD);
  float s2 = 0.f;
  for (int c = 0; c < 32; ++c) { float d = vals[c] - mu; s2 += d * d; }
  for (int m = 1; m < 8; m <<= 1) s2 += __shfl_xor(s2, m, 32);
  const float rstd = rsqrtf(s2 * (1.f / ENCD) + 1e-5f);
  for (int c = 0; c < 32; ++c) {
    const int col = j * 32 + c;
    float y = (vals[c] - mu) * rstd * g[col] + b[col];
    xres[row * ENCD + col] = y;
    xb[row * ENCD + col]   = f2bf(y);
  }
}

// ---------------- weight convert+transpose: out[N][K] = bf16(in[K][N]) ------
__global__ void transpose_bf16_kernel(const float* __restrict__ in,
                                      unsigned short* __restrict__ out,
                                      int K, int N) {
  int i = blockIdx.x * blockDim.x + threadIdx.x;
  if (i >= K * N) return;
  int n = i / K, k = i % K;
  out[i] = f2bf(in[k * N + n]);
}

// ---------------- fused set-encoder: one block (8 waves) per node -----------
__global__ void encoder_kernel(const float* __restrict__ posts,
                               const unsigned short* __restrict__ projT,
                               const float* __restrict__ proj_b,
                               const unsigned short* __restrict__ qkvT,
                               const float* __restrict__ qkv_b,
                               const unsigned short* __restrict__ outT,
                               const float* __restrict__ out_b,
                               const float* __restrict__ ln1_g,
                               const float* __restrict__ ln1_b,
                               const unsigned short* __restrict__ ff1T,
                               const float* __restrict__ ff1_b,
                               const unsigned short* __restrict__ ff2T,
                               const float* __restrict__ ff2_b,
                               const float* __restrict__ ln2_g,
                               const float* __restrict__ ln2_b,
                               float* __restrict__ hout) {
  extern __shared__ __align__(16) char smem[];
  // LDS pool layout (bytes)
  unsigned short* P    = (unsigned short*)(smem);             // 32x512 bf16 / reused as F 32x1024
  unsigned short* Fb   = (unsigned short*)(smem);             // 64 KB region
  float*          xf   = (float*)(smem + 65536);              // 32x256 f32
  unsigned short* xb   = (unsigned short*)(smem + 98304);     // 32x256 bf16
  unsigned short* qkv  = (unsigned short*)(smem + 114688);    // 32x768 bf16
  unsigned short* vT   = (unsigned short*)(smem + 163840);    // 256x32 bf16
  float*          sc   = (float*)(smem + 180224);             // 4x32x32 f32
  unsigned short* attn = (unsigned short*)(smem + 196608);    // 4x32x32 bf16
  float*          acc  = (float*)(smem + 200704);             // 32x256 f32

  const int tid = threadIdx.x, lane = tid & 31, wave = tid >> 5;
  const int n = blockIdx.x;
  const float* pn = posts + (size_t)n * (SEQ * PDIM);

  // ---- stage 1: posts -> bf16 LDS
  for (int i4 = tid; i4 < (SEQ * PDIM) / 4; i4 += 256) {
    float4 v = ((const float4*)pn)[i4];
    P[i4 * 4 + 0] = f2bf(v.x); P[i4 * 4 + 1] = f2bf(v.y);
    P[i4 * 4 + 2] = f2bf(v.z); P[i4 * 4 + 3] = f2bf(v.w);
  }
  __syncthreads();

  // ---- stage 2: x = P @ proj + b   (f32 in xf, bf16 in xb)
  gemm_bf16<PDIM>(P, PDIM, projT, PDIM, proj_b, xf, ENCD, xb, ENCD, 2, 16, false, tid);
  __syncthreads();

  // ---- stage 3: qkv = x @ qkv_w + b  (bf16, [s][768])
  gemm_bf16<ENCD>(xb, ENCD, qkvT, ENCD, qkv_b, nullptr, 0, qkv, 768, 2, 48, false, tid);
  __syncthreads();

  // ---- stage 4: vT[dh][s] = v[s][dh]
  for (int i = tid; i < ENCD * SEQ; i += 256) {
    int dh = i >> 5, s = i & 31;
    vT[i] = qkv[s * 768 + 512 + dh];
  }
  __syncthreads();

  // ---- stage 5: scores[h] = Q_h @ K_h^T   (WMMA, K=64)
  {
    const int khalf = (lane & 16) ? 8 : 0, klo = (lane & 16) ? 8 : 0;
    for (int t = wave; t < 16; t += 8) {
      int h = t >> 2, mt = t & 1, nt = (t >> 1) & 1;
      const unsigned short* Aq = qkv + h * DH;
      const unsigned short* Bk = qkv + 256 + h * DH;
      int m = mt * 16 + (lane & 15), nB = nt * 16 + (lane & 15);
      v8f c = {};
      for (int k0 = 0; k0 < DH; k0 += 32) {
        v16bf a = load_frag(Aq + m * 768 + k0 + khalf, Aq + m * 768 + k0 + 16 + khalf);
        v16bf b = load_frag(Bk + nB * 768 + k0 + klo,  Bk + nB * 768 + k0 + 16 + klo);
        c = wmma_bf16(a, b, c);
      }
      float* S = sc + h * 1024;
      int nn = nt * 16 + (lane & 15), mr0 = mt * 16 + ((lane & 16) ? 8 : 0);
      for (int r = 0; r < 8; ++r) S[(mr0 + r) * 32 + nn] = c[r];
    }
  }
  __syncthreads();

  // ---- stage 6: softmax (scale 1/sqrt(64)); pad_mask is all-false
  if (tid < 128) {
    int h = tid >> 5, q = tid & 31;
    const float* S = sc + h * 1024 + q * 32;
    float mx = -1e30f;
    for (int k = 0; k < 32; ++k) mx = fmaxf(mx, S[k] * 0.125f);
    float sum = 0.f; float e[32];
    for (int k = 0; k < 32; ++k) { e[k] = __expf(S[k] * 0.125f - mx); sum += e[k]; }
    float inv = 1.f / sum;
    for (int k = 0; k < 32; ++k) attn[h * 1024 + q * 32 + k] = f2bf(e[k] * inv);
  }
  __syncthreads();

  // ---- stage 7: O = attn @ V  (WMMA, K=32) -> acc f32 + xb bf16
  {
    const int khalf = (lane & 16) ? 8 : 0, klo = (lane & 16) ? 8 : 0;
    for (int t = wave; t < 32; t += 8) {
      int h = t >> 3, r8 = t & 7, mt = r8 & 1, nt = r8 >> 1;  // nt 0..3 over DH
      const unsigned short* Aa = attn + h * 1024;
      const unsigned short* Bv = vT + h * DH * 32;
      int m = mt * 16 + (lane & 15), nB = nt * 16 + (lane & 15);
      v16bf a = load_frag(Aa + m * 32 + khalf, Aa + m * 32 + 16 + khalf);
      v16bf b = load_frag(Bv + nB * 32 + klo,  Bv + nB * 32 + 16 + klo);
      v8f c = {};
      c = wmma_bf16(a, b, c);
      int ng = h * DH + nt * 16 + (lane & 15);
      int mr0 = mt * 16 + ((lane & 16) ? 8 : 0);
      for (int r = 0; r < 8; ++r) {
        float v = c[r];
        acc[(mr0 + r) * ENCD + ng] = v;
        xb[(mr0 + r) * ENCD + ng]  = f2bf(v);
      }
    }
  }
  __syncthreads();

  // ---- stage 8: out-proj: acc = O @ out_w + b
  gemm_bf16<ENCD>(xb, ENCD, outT, ENCD, out_b, acc, ENCD, nullptr, 0, 2, 16, false, tid);
  __syncthreads();

  // ---- stage 9: LN1(x + o) -> xf, xb
  ln_residual(xf, acc, ln1_g, ln1_b, xb, tid);
  __syncthreads();

  // ---- stage 10: F = relu(x @ ff1 + b)  (bf16, 32x1024)
  gemm_bf16<ENCD>(xb, ENCD, ff1T, ENCD, ff1_b, nullptr, 0, Fb, FFD, 2, 64, true, tid);
  __syncthreads();

  // ---- stage 11: acc = F @ ff2 + b
  gemm_bf16<FFD>(Fb, FFD, ff2T, FFD, ff2_b, acc, ENCD, nullptr, 0, 2, 16, false, tid);
  __syncthreads();

  // ---- stage 12: LN2(x + f) -> xf
  ln_residual(xf, acc, ln2_g, ln2_b, xb, tid);
  __syncthreads();

  // ---- stage 13: masked mean over S (mask all-valid) -> h[n]
  float s = 0.f;
  for (int r = 0; r < SEQ; ++r) s += xf[r * ENCD + tid];
  hout[(size_t)n * ENCD + tid] = s * (1.f / SEQ);
}

// ---------------- GraphSAGE layer: 16 nodes per block ------------------------
__global__ void gnn_kernel(const float* __restrict__ hin,
                           const int* __restrict__ neighbors,
                           const unsigned short* __restrict__ gT,   // [256][512]
                           const float* __restrict__ gbias,
                           float* __restrict__ hout) {
  __shared__ unsigned short A[16 * 512];   // [node][ self(256) | mean(256) ] bf16
  __shared__ int nbs[16 * KNB];
  const int tid = threadIdx.x;
  const int n0 = blockIdx.x * 16;

  for (int i = tid; i < 16 * KNB; i += 256)
    nbs[i] = neighbors[(size_t)(n0 + (i >> 5)) * KNB + (i & 31)];
  __syncthreads();

  for (int i = tid; i < 16 * ENCD; i += 256) {
    int node = i >> 8, c = i & 255;
    A[node * 512 + c] = f2bf(hin[(size_t)(n0 + node) * ENCD + c]);
    float s = 0.f;
    for (int k = 0; k < KNB; ++k)
      s += hin[(size_t)nbs[node * KNB + k] * ENCD + c];   // L2-resident gathers
    A[node * 512 + ENCD + c] = f2bf(s * (1.f / KNB));
  }
  __syncthreads();

  gemm_bf16<512>(A, 512, gT, 512, gbias, hout + (size_t)n0 * ENCD, ENCD,
                 nullptr, 0, 1, 16, true, tid);
}

// ---------------- host side --------------------------------------------------
extern "C" void kernel_launch(void* const* d_in, const int* in_sizes, int n_in,
                              void* d_out, int out_size, void* d_ws, size_t ws_size,
                              hipStream_t stream) {
  const float* posts  = (const float*)d_in[0];
  const int*   neigh  = (const int*)d_in[2];
  const float* proj_w = (const float*)d_in[3];  const float* proj_b = (const float*)d_in[4];
  const float* qkv_w  = (const float*)d_in[5];  const float* qkv_b  = (const float*)d_in[6];
  const float* out_w  = (const float*)d_in[7];  const float* out_b  = (const float*)d_in[8];
  const float* ln1_g  = (const float*)d_in[9];  const float* ln1_b  = (const float*)d_in[10];
  const float* ff1_w  = (const float*)d_in[11]; const float* ff1_b  = (const float*)d_in[12];
  const float* ff2_w  = (const float*)d_in[13]; const float* ff2_b  = (const float*)d_in[14];
  const float* ln2_g  = (const float*)d_in[15]; const float* ln2_b  = (const float*)d_in[16];
  const float* gnn_w  = (const float*)d_in[17]; const float* gnn_b  = (const float*)d_in[18];

  // workspace carve-up
  char* ws = (char*)d_ws;
  size_t off = 0;
  auto carve = [&](size_t bytes) { char* p = ws + off; off = (off + bytes + 255) & ~(size_t)255; return p; };
  unsigned short* projT = (unsigned short*)carve((size_t)PDIM * ENCD * 2);
  unsigned short* qkvT  = (unsigned short*)carve((size_t)ENCD * 768 * 2);
  unsigned short* outT  = (unsigned short*)carve((size_t)ENCD * ENCD * 2);
  unsigned short* ff1T  = (unsigned short*)carve((size_t)ENCD * FFD * 2);
  unsigned short* ff2T  = (unsigned short*)carve((size_t)FFD * ENCD * 2);
  unsigned short* gT0   = (unsigned short*)carve((size_t)512 * ENCD * 2);
  unsigned short* gT1   = (unsigned short*)carve((size_t)512 * ENCD * 2);
  float* h0 = (float*)carve((size_t)N_NODES * ENCD * 4);
  float* h1 = (float*)carve((size_t)N_NODES * ENCD * 4);

  auto tlaunch = [&](const float* in, unsigned short* out, int K, int N) {
    int total = K * N;
    transpose_bf16_kernel<<<(total + 255) / 256, 256, 0, stream>>>(in, out, K, N);
  };
  tlaunch(proj_w, projT, PDIM, ENCD);
  tlaunch(qkv_w,  qkvT,  ENCD, 768);
  tlaunch(out_w,  outT,  ENCD, ENCD);
  tlaunch(ff1_w,  ff1T,  ENCD, FFD);
  tlaunch(ff2_w,  ff2T,  FFD,  ENCD);
  tlaunch(gnn_w,            gT0, 512, ENCD);
  tlaunch(gnn_w + 512*ENCD, gT1, 512, ENCD);

  const size_t enc_lds = 233472;  // 228 KB dynamic LDS (<= 320 KB/WG on CDNA5)
  encoder_kernel<<<N_NODES, 256, enc_lds, stream>>>(
      posts, projT, proj_b, qkvT, qkv_b, outT, out_b,
      ln1_g, ln1_b, ff1T, ff1_b, ff2T, ff2_b, ln2_g, ln2_b, h0);

  gnn_kernel<<<N_NODES / 16, 256, 0, stream>>>(h0, neigh, gT0, gnn_b,        h1);
  gnn_kernel<<<N_NODES / 16, 256, 0, stream>>>(h1, neigh, gT1, gnn_b + ENCD, (float*)d_out);
}
